// MaxPoolingAggregator_31507880083682
// MI455X (gfx1250) — matrix operand order repaired and checked by reading, not compiled, not run on hardware
//
#include <hip/hip_runtime.h>

typedef __attribute__((ext_vector_type(16))) _Float16 v16h;
typedef __attribute__((ext_vector_type(8)))  _Float16 h8;
typedef __attribute__((ext_vector_type(8)))  float    f8;
typedef __attribute__((ext_vector_type(8)))  float    v8f;

#define DDIM   512
#define NNEI   32
#define BATCH  8192
#define WELEMS (DDIM * DDIM)   // 262144 elements per weight matrix

// ---------------------------------------------------------------------------
// Kernel 1: convert fp32 weights -> f16, transposed to n-major [n][k] layout
// so WMMA B-fragments are contiguous 32B per lane.
// ---------------------------------------------------------------------------
__global__ __launch_bounds__(256) void prep_weights(
    const float* __restrict__ mlp_w, const float* __restrict__ neigh_w,
    const float* __restrict__ self_w,
    _Float16* __restrict__ wTa, _Float16* __restrict__ wTn,
    _Float16* __restrict__ wTs)
{
    int idx = blockIdx.x * 256 + threadIdx.x;      // 0 .. 3*WELEMS-1
    int m   = idx >> 18;                           // which matrix
    int rem = idx & (WELEMS - 1);
    int k   = rem & (DDIM - 1);                    // contiguous in dst
    int n   = rem >> 9;
    const float* src = (m == 0) ? mlp_w : (m == 1) ? neigh_w : self_w;
    _Float16*    dst = (m == 0) ? wTa   : (m == 1) ? wTn     : wTs;
    dst[(size_t)n * DDIM + k] = (_Float16)src[(size_t)k * DDIM + n];
}

// Build a WMMA A/B v16h fragment from two 8-wide chunks.
__device__ inline v16h pack16(h8 lo, h8 hi) {
    v16h t;
#pragma unroll
    for (int i = 0; i < 8; ++i) { t[i] = lo[i]; t[8 + i] = hi[i]; }
    return t;
}
__device__ inline v16h cvt16(f8 lo, f8 hi) {
    v16h t;
#pragma unroll
    for (int i = 0; i < 8; ++i) { t[i] = (_Float16)lo[i]; t[8 + i] = (_Float16)hi[i]; }
    return t;
}

// ---------------------------------------------------------------------------
// Kernel 2: h = neigh_vecs @ mlp_w ; pooled = relu(max_n(h) + b)
// One workgroup per neighbor group b (32 rows). 8 waves, each wave owns
// 64 output columns and does the full 32-row max-pool internally.
// ---------------------------------------------------------------------------
__global__ __launch_bounds__(256) void gemm1_pool(
    const float* __restrict__ neigh, const _Float16* __restrict__ wT,
    const float* __restrict__ bias, _Float16* __restrict__ pooled)
{
    const int b    = blockIdx.x;
    const int tid  = threadIdx.x;
    const int wave = tid >> 5;
    const int lane = tid & 31;
    const int lh   = lane & 15;
    const bool hi  = lane >= 16;
    const int nbase = wave * 64;

    const float* Abase = neigh + (size_t)b * NNEI * DDIM;

    v8f acc[2][4];
#pragma unroll
    for (int i = 0; i < 2; ++i)
#pragma unroll
        for (int j = 0; j < 4; ++j) acc[i][j] = (v8f)(0.0f);

    for (int k0 = 0; k0 < DDIM; k0 += 32) {
        // --- A fragments (16x32 f16 layout): two m-fragments of 16 rows ---
        v16h afrag[2];
#pragma unroll
        for (int fm = 0; fm < 2; ++fm) {
            const float* rp = Abase + (size_t)(fm * 16 + lh) * DDIM
                              + k0 + (hi ? 16 : 0);
            f8 c0 = *(const f8*)rp;
            f8 c1 = *(const f8*)(rp + 8);
            afrag[fm] = cvt16(c0, c1);
        }
        // prefetch next A k-panel while WMMAs run
        if (k0 + 32 < DDIM)
            __builtin_prefetch(Abase + (size_t)lh * DDIM + k0 + 32, 0, 1);

        // --- B fragments: contiguous 32B per lane from n-major wT ---
#pragma unroll
        for (int fn = 0; fn < 4; ++fn) {
            const _Float16* bp = wT + (size_t)(nbase + fn * 16 + lh) * DDIM
                                 + k0 + (hi ? 16 : 0);
            v16h bfrag = *(const v16h*)bp;
            acc[0][fn] = __builtin_amdgcn_wmma_f32_16x16x32_f16(
                false, afrag[0], false, bfrag, (short)0, acc[0][fn], false, false);
            acc[1][fn] = __builtin_amdgcn_wmma_f32_16x16x32_f16(
                false, afrag[1], false, bfrag, (short)0, acc[1][fn], false, false);
        }
    }

    // --- max-pool over the 32 rows, then bias + relu (monotone => commute) ---
#pragma unroll
    for (int fn = 0; fn < 4; ++fn) {
        float m = acc[0][fn][0];
#pragma unroll
        for (int r = 1; r < 8; ++r) m = fmaxf(m, acc[0][fn][r]);
#pragma unroll
        for (int r = 0; r < 8; ++r) m = fmaxf(m, acc[1][fn][r]);
        // merge the other 16 rows held by the opposite lane half (xor 16)
        int other = __builtin_amdgcn_ds_swizzle(__float_as_int(m), 0x401F); // SWAPX16
        m = fmaxf(m, __int_as_float(other));
        int n = nbase + fn * 16 + lh;
        float v = m + bias[n];
        v = v > 0.0f ? v : 0.0f;
        if (!hi) pooled[(size_t)b * DDIM + n] = (_Float16)v;
    }
}

// ---------------------------------------------------------------------------
// Kernel 3: out = relu(pooled @ neigh_w + self_vecs @ self_w)
// One workgroup per 32 output rows; 8 waves x 64 columns.
// ---------------------------------------------------------------------------
__global__ __launch_bounds__(256) void gemm2_out(
    const _Float16* __restrict__ pooled, const _Float16* __restrict__ wTn,
    const float* __restrict__ selfv, const _Float16* __restrict__ wTs,
    float* __restrict__ out)
{
    const int mb   = blockIdx.x;          // 32-row block
    const int tid  = threadIdx.x;
    const int wave = tid >> 5;
    const int lane = tid & 31;
    const int lh   = lane & 15;
    const bool hi  = lane >= 16;
    const int nbase = wave * 64;
    const int rowBase = mb * 32;

    v8f acc[2][4];
#pragma unroll
    for (int i = 0; i < 2; ++i)
#pragma unroll
        for (int j = 0; j < 4; ++j) acc[i][j] = (v8f)(0.0f);

    // ---- pass 1: pooled (f16, direct loads) x neigh_w ----
    for (int k0 = 0; k0 < DDIM; k0 += 32) {
        v16h afrag[2];
#pragma unroll
        for (int fm = 0; fm < 2; ++fm) {
            const _Float16* rp = pooled + (size_t)(rowBase + fm * 16 + lh) * DDIM
                                 + k0 + (hi ? 16 : 0);
            afrag[fm] = pack16(*(const h8*)rp, *(const h8*)(rp + 8));
        }
#pragma unroll
        for (int fn = 0; fn < 4; ++fn) {
            const _Float16* bp = wTn + (size_t)(nbase + fn * 16 + lh) * DDIM
                                 + k0 + (hi ? 16 : 0);
            v16h bfrag = *(const v16h*)bp;
            acc[0][fn] = __builtin_amdgcn_wmma_f32_16x16x32_f16(
                false, afrag[0], false, bfrag, (short)0, acc[0][fn], false, false);
            acc[1][fn] = __builtin_amdgcn_wmma_f32_16x16x32_f16(
                false, afrag[1], false, bfrag, (short)0, acc[1][fn], false, false);
        }
    }

    // ---- pass 2: self_vecs (fp32 -> f16 on the fly) x self_w ----
    for (int k0 = 0; k0 < DDIM; k0 += 32) {
        v16h afrag[2];
#pragma unroll
        for (int fm = 0; fm < 2; ++fm) {
            const float* rp = selfv + (size_t)(rowBase + fm * 16 + lh) * DDIM
                              + k0 + (hi ? 16 : 0);
            afrag[fm] = cvt16(*(const f8*)rp, *(const f8*)(rp + 8));
        }
#pragma unroll
        for (int fn = 0; fn < 4; ++fn) {
            const _Float16* bp = wTs + (size_t)(nbase + fn * 16 + lh) * DDIM
                                 + k0 + (hi ? 16 : 0);
            v16h bfrag = *(const v16h*)bp;
            acc[0][fn] = __builtin_amdgcn_wmma_f32_16x16x32_f16(
                false, afrag[0], false, bfrag, (short)0, acc[0][fn], false, false);
            acc[1][fn] = __builtin_amdgcn_wmma_f32_16x16x32_f16(
                false, afrag[1], false, bfrag, (short)0, acc[1][fn], false, false);
        }
    }

    // ---- epilogue: relu, store fp32 ----
#pragma unroll
    for (int fm = 0; fm < 2; ++fm)
#pragma unroll
        for (int fn = 0; fn < 4; ++fn) {
            int col = nbase + fn * 16 + lh;
#pragma unroll
            for (int r = 0; r < 8; ++r) {
                int row = rowBase + fm * 16 + r + (hi ? 8 : 0);
                float v = acc[fm][fn][r];
                out[(size_t)row * DDIM + col] = v > 0.0f ? v : 0.0f;
            }
        }
}

// ---------------------------------------------------------------------------
extern "C" void kernel_launch(void* const* d_in, const int* in_sizes, int n_in,
                              void* d_out, int out_size, void* d_ws, size_t ws_size,
                              hipStream_t stream) {
    const float* self_vecs  = (const float*)d_in[0];
    const float* neigh_vecs = (const float*)d_in[1];
    const float* mlp_w      = (const float*)d_in[2];
    const float* mlp_b      = (const float*)d_in[3];
    const float* neigh_w    = (const float*)d_in[4];
    const float* self_w     = (const float*)d_in[5];
    float*       out        = (float*)d_out;

    _Float16* wTa    = (_Float16*)d_ws;                 // 512 KB
    _Float16* wTn    = wTa + WELEMS;                    // 512 KB
    _Float16* wTs    = wTn + WELEMS;                    // 512 KB
    _Float16* pooled = wTs + WELEMS;                    // 8 MB

    prep_weights<<<(3 * WELEMS) / 256, 256, 0, stream>>>(
        mlp_w, neigh_w, self_w, wTa, wTn, wTs);

    gemm1_pool<<<BATCH, 256, 0, stream>>>(neigh_vecs, wTa, mlp_b, pooled);

    gemm2_out<<<BATCH / 32, 256, 0, stream>>>(pooled, wTn, self_vecs, wTs, out);
}
